// QuantizedLinear_15006615733607
// MI455X (gfx1250) — 8-bit matrix operand order repaired and checked by reading, compiled, not run on hardware
//
#include <hip/hip_runtime.h>

typedef __attribute__((ext_vector_type(8))) int v8i;
typedef __attribute__((ext_vector_type(4))) int v4i;
typedef __attribute__((ext_vector_type(2))) int v2i;

#define MDIM 8192
#define NDIM 4096
#define KDIM 4096
#define BM   128
#define BN   128
#define BK   64
#define LDP  80

#if __has_builtin(__builtin_amdgcn_global_load_async_to_lds_b128)
#define HAVE_ASYNC_LDS 1
#else
#define HAVE_ASYNC_LDS 0
#endif

#if HAVE_ASYNC_LDS
typedef __attribute__((address_space(1))) v4i as1_v4i;
typedef __attribute__((address_space(3))) v4i as3_v4i;
__device__ __forceinline__ void gasync_b128(const signed char* gsrc, signed char* ldst) {
    __builtin_amdgcn_global_load_async_to_lds_b128(
        (as1_v4i*)gsrc, (as3_v4i*)ldst,  0,  0);
}
#if __has_builtin(__builtin_amdgcn_s_wait_asynccnt)
#define WAIT_ASYNC() __builtin_amdgcn_s_wait_asynccnt(0)
#else
#define WAIT_ASYNC() asm volatile("s_wait_asynccnt 0x0" ::: "memory")
#endif
#endif


__global__ __launch_bounds__(256) void quant_x_kernel(const float* __restrict__ x,
                                                      const float* __restrict__ iscale,
                                                      signed char* __restrict__ xq) {
    const float s = iscale[0];
    const size_t base = ((size_t)blockIdx.x * blockDim.x + threadIdx.x) * 4;
    const float4 v = *(const float4*)(x + base);
    int t0 = (int)(v.x * s);
    int t1 = (int)(v.y * s);
    int t2 = (int)(v.z * s);
    int t3 = (int)(v.w * s);
    t0 = t0 < -128 ? -128 : (t0 > 127 ? 127 : t0);
    t1 = t1 < -128 ? -128 : (t1 > 127 ? 127 : t1);
    t2 = t2 < -128 ? -128 : (t2 > 127 ? 127 : t2);
    t3 = t3 < -128 ? -128 : (t3 > 127 ? 127 : t3);
    char4 q;
    q.x = (signed char)t0; q.y = (signed char)t1;
    q.z = (signed char)t2; q.w = (signed char)t3;
    *(char4*)(xq + base) = q;
}

__global__ __launch_bounds__(256) void quant_w_kernel(const float* __restrict__ w,
                                                      const float* __restrict__ ws,
                                                      signed char* __restrict__ wq) {
    const size_t base = ((size_t)blockIdx.x * blockDim.x + threadIdx.x) * 4;
    const float s = ws[base >> 12];
    const float4 v = *(const float4*)(w + base);
    char4 q;
    q.x = (signed char)(int)rintf(v.x * s);
    q.y = (signed char)(int)rintf(v.y * s);
    q.z = (signed char)(int)rintf(v.z * s);
    q.w = (signed char)(int)rintf(v.w * s);
    *(char4*)(wq + base) = q;
}


__global__ __launch_bounds__(256) void qgemm_i8_wmma(
    const signed char* __restrict__ Aq,
    const signed char* __restrict__ Wq,
    const float* __restrict__ bias,
    const float* __restrict__ wscale,
    const float* __restrict__ iscale,
    float* __restrict__ out)
{
    __shared__ signed char lds_a[2][BM * LDP];
    __shared__ signed char lds_b[2][BN * LDP];

    const int tid  = threadIdx.x;
    const int lane = tid & 31;
    const int wave = tid >> 5;
    const int block_m = blockIdx.y * BM;
    const int block_n = blockIdx.x * BN;

    const int lrow0 = tid >> 2;
    const int lrow1 = lrow0 + 64;
    const int lkoff = (tid & 3) << 4;

    const signed char* gA = Aq + (size_t)block_m * KDIM + lkoff;
    const signed char* gB = Wq + (size_t)block_n * KDIM + lkoff;
    const int ldsoff0 = lrow0 * LDP + lkoff;
    const int ldsoff1 = lrow1 * LDP + lkoff;

    v8i acc[8] = {};

    signed char* cura = &lds_a[0][0];
    signed char* curb = &lds_b[0][0];
    signed char* nxta = &lds_a[1][0];
    signed char* nxtb = &lds_b[1][0];

#if HAVE_ASYNC_LDS
    gasync_b128(gA + (size_t)lrow0 * KDIM, cura + ldsoff0);
    gasync_b128(gA + (size_t)lrow1 * KDIM, cura + ldsoff1);
    gasync_b128(gB + (size_t)lrow0 * KDIM, curb + ldsoff0);
    gasync_b128(gB + (size_t)lrow1 * KDIM, curb + ldsoff1);
    WAIT_ASYNC();
    __syncthreads();
#else
    {
        v4i ra0 = *(const v4i*)(gA + (size_t)lrow0 * KDIM);
        v4i ra1 = *(const v4i*)(gA + (size_t)lrow1 * KDIM);
        v4i rb0 = *(const v4i*)(gB + (size_t)lrow0 * KDIM);
        v4i rb1 = *(const v4i*)(gB + (size_t)lrow1 * KDIM);
        *(v4i*)(cura + ldsoff0) = ra0;
        *(v4i*)(cura + ldsoff1) = ra1;
        *(v4i*)(curb + ldsoff0) = rb0;
        *(v4i*)(curb + ldsoff1) = rb1;
    }
    __syncthreads();
#endif

    const int aoff = ((wave << 4) + (lane & 15)) * LDP + (((lane >> 4)) << 4);
    const int boff = (lane & 15) * LDP + ((lane >> 4) << 4);

    const int NKB = KDIM / BK;
    #pragma unroll 1
    for (int kb = 0; kb < NKB; ++kb) {
        const bool more = (kb + 1) < NKB;

#if HAVE_ASYNC_LDS
        if (more) {
            const size_t go = (size_t)(kb + 1) * BK;
            gasync_b128(gA + (size_t)lrow0 * KDIM + go, nxta + ldsoff0);
            gasync_b128(gA + (size_t)lrow1 * KDIM + go, nxta + ldsoff1);
            gasync_b128(gB + (size_t)lrow0 * KDIM + go, nxtb + ldsoff0);
            gasync_b128(gB + (size_t)lrow1 * KDIM + go, nxtb + ldsoff1);
        }
#else
        v4i ra0, ra1, rb0, rb1;
        if (more) {
            const size_t go = (size_t)(kb + 1) * BK;
            ra0 = *(const v4i*)(gA + (size_t)lrow0 * KDIM + go);
            ra1 = *(const v4i*)(gA + (size_t)lrow1 * KDIM + go);
            rb0 = *(const v4i*)(gB + (size_t)lrow0 * KDIM + go);
            rb1 = *(const v4i*)(gB + (size_t)lrow1 * KDIM + go);
        }
#endif

        const signed char* abase = cura + aoff;
        v2i a01 = *(const v2i*)(abase + 0);
        v2i a23 = *(const v2i*)(abase + 8);
        v2i a45 = *(const v2i*)(abase + 32);
        v2i a67 = *(const v2i*)(abase + 40);
        v8i afrag = { a01[0], a01[1], a23[0], a23[1],
                      a45[0], a45[1], a67[0], a67[1] };

        v8i bfrag[8];
        #pragma unroll
        for (int t = 0; t < 8; ++t) {
            const signed char* bbase = curb + boff + (t << 4) * LDP;
            v4i blo = *(const v4i*)(bbase + 0);
            v4i bhi = *(const v4i*)(bbase + 32);
            bfrag[t] = (v8i){ blo[0], blo[1], blo[2], blo[3],
                              bhi[0], bhi[1], bhi[2], bhi[3] };
        }

        #pragma unroll
        for (int t = 0; t < 8; ++t) {
            acc[t] = __builtin_amdgcn_wmma_i32_16x16x64_iu8(
                         true, afrag, true, bfrag[t], acc[t], false, false);
        }

#if HAVE_ASYNC_LDS
        if (more) WAIT_ASYNC();
#else
        if (more) {
            *(v4i*)(nxta + ldsoff0) = ra0;
            *(v4i*)(nxta + ldsoff1) = ra1;
            *(v4i*)(nxtb + ldsoff0) = rb0;
            *(v4i*)(nxtb + ldsoff1) = rb1;
        }
#endif
        __syncthreads();

        signed char* ta = cura; cura = nxta; nxta = ta;
        signed char* tb = curb; curb = nxtb; nxtb = tb;
    }

    const float is = iscale[0];
    const int m0 = block_m + (wave << 4) + ((lane >> 4) << 3);
    #pragma unroll
    for (int t = 0; t < 8; ++t) {
        const int n = block_n + (t << 4) + (lane & 15);
        const float inv = 1.0f / (wscale[n] * is);
        const float bv  = bias[n];
        float* po = out + (size_t)m0 * NDIM + n;
        #pragma unroll
        for (int v = 0; v < 8; ++v) {
            po[(size_t)v * NDIM] = ((float)acc[t][v] + bv) * inv;
        }
    }
}


extern "C" void kernel_launch(void* const* d_in, const int* in_sizes, int n_in,
                              void* d_out, int out_size, void* d_ws, size_t ws_size,
                              hipStream_t stream) {
    const float* x      = (const float*)d_in[0];
    const float* w      = (const float*)d_in[1];
    const float* bias   = (const float*)d_in[2];
    const float* wscale = (const float*)d_in[3];
    const float* iscale = (const float*)d_in[4];
    float* out = (float*)d_out;

    signed char* xq = (signed char*)d_ws;
    signed char* wq = xq + (size_t)MDIM * KDIM;

    quant_x_kernel<<<(MDIM * (size_t)KDIM) / 4 / 256, 256, 0, stream>>>(x, iscale, xq);
    quant_w_kernel<<<(NDIM * (size_t)KDIM) / 4 / 256, 256, 0, stream>>>(w, wscale, wq);

    dim3 grid(NDIM / BN, MDIM / BM);
    qgemm_i8_wmma<<<grid, 256, 0, stream>>>(xq, wq, bias, wscale, iscale, out);
}
